// Transformer_40510131536392
// MI455X (gfx1250) — matrix operand order repaired and checked by reading, not compiled, not run on hardware
//
#include <hip/hip_runtime.h>
#include <hip/hip_bf16.h>
#include <math.h>

// ---------------------------------------------------------------------------
// Model constants (match the reference)
// ---------------------------------------------------------------------------
constexpr int Vv  = 32000;
constexpr int Dm  = 1024;
constexpr int HFF = 4096;
constexpr int NHh = 16;
constexpr int LEe = 6;
constexpr int LDd = 6;
constexpr int Bb  = 2;
constexpr int Ss  = 1024;          // SS == ST == 1024
constexpr int DH  = Dm / NHh;      // 64
constexpr int TOK = Bb * Ss;       // 2048 rows

// ---------------------------------------------------------------------------
// bf16 / WMMA vector types (CDNA5 wave32)
// ---------------------------------------------------------------------------
typedef __bf16 bf16_t;
typedef __attribute__((ext_vector_type(16))) __bf16 v16bf;
typedef __attribute__((ext_vector_type(8)))  __bf16 v8bf;
typedef __attribute__((ext_vector_type(4)))  __bf16 v4bf;
typedef __attribute__((ext_vector_type(8)))  float  v8f;

__device__ __forceinline__ float gelu_erf(float x) {
    return 0.5f * x * (1.0f + erff(x * 0.70710678118654752440f));
}

__device__ __forceinline__ v4bf pack4_bf16(float4 f) {
    v4bf p;
    p.x = (bf16_t)f.x; p.y = (bf16_t)f.y;
    p.z = (bf16_t)f.z; p.w = (bf16_t)f.w;
    return p;
}

// ---------------------------------------------------------------------------
// General batched GEMM:  C[z] = act(A[z] @ B[z] + bias) + Res[z]
//   A: M x K (fp32, row-major, leading dim lda)
//   B: K x N (bT=0) or stored as N x K (bT=1) row-major, leading dim ldb
//   batch offset for operand X: (z/nh)*sX2 + (z%nh)*sX1
//
// fp32 inputs converted to bf16 at LDS-store time (packed ds_store_b64);
// f32 accumulation via v_wmma_f32_16x16x32_bf16. 256 threads = 8 waves;
// block tile 128x128, K step 32, each wave owns 4x2 16x16 accumulators.
// Software pipeline: raw float4 tiles for step k+1 are issued before the
// WMMA block of step k, so s_wait_loadcnt hides behind matrix compute.
// Interior tiles take an unguarded fast path; edges use per-element guards.
// ---------------------------------------------------------------------------
#define BM 128
#define BN 128
#define BK 32
#define LDSS 40   // bf16 elems per LDS row: 80B, 16B-aligned, conflict-free

__global__ __launch_bounds__(256)
void gemm_bf16_wmma(const float* __restrict__ A, int lda, long long sA1, long long sA2,
                    const float* __restrict__ B, int ldb, long long sB1, long long sB2, int bT,
                    const float* __restrict__ bias,
                    const float* __restrict__ Res, int ldr,
                    float* __restrict__ C, int ldc, long long sC1, long long sC2,
                    int M, int N, int K, int nh, int act)
{
    __shared__ alignas(16) bf16_t As[BM * LDSS];
    __shared__ alignas(16) bf16_t Bs[BN * LDSS];

    const int z  = blockIdx.z;
    const int zb = z / nh, zh = z % nh;
    A += (long long)zb * sA2 + (long long)zh * sA1;
    B += (long long)zb * sB2 + (long long)zh * sB1;
    C += (long long)zb * sC2 + (long long)zh * sC1;
    if (Res) Res += (long long)zb * sC2 + (long long)zh * sC1;

    const int row0 = blockIdx.y * BM;
    const int col0 = blockIdx.x * BN;
    const int tid  = threadIdx.x;
    const int lane = tid & 31;
    const int wave = tid >> 5;
    const int wm   = wave & 1;   // 0..1 -> 64-row half
    const int wn   = wave >> 1;  // 0..3 -> 32-col quarter

    const bool mFull = (row0 + BM <= M);
    const bool nFull = (col0 + BN <= N);

    // staging thread coordinates
    const int sm  = tid >> 3;          // 0..31: A (and B^T) row base
    const int sk4 = (tid & 7) << 2;    // 0..28: k offset (x4)
    const int skb = tid >> 5;          // 0..7 : B k base (bT=0)
    const int snb = (tid & 31) << 2;   // 0..124: B n offset (bT=0)

    float4 ra[4], rb[4];

    // ---- issue global loads for the A tile at k0 into registers ----
    auto loadA = [&](int k0) {
        const bool fast = mFull && (k0 + BK <= K);
        #pragma unroll
        for (int r = 0; r < 4; ++r) {
            const int gm = row0 + sm + (r << 5);
            const int gk = k0 + sk4;
            if (fast) {
                ra[r] = *(const float4*)(A + (long long)gm * lda + gk);
            } else {
                float4 t = {0.f, 0.f, 0.f, 0.f};
                if (gm < M) {
                    const float* ap = A + (long long)gm * lda + gk;
                    if (gk     < K) t.x = ap[0];
                    if (gk + 1 < K) t.y = ap[1];
                    if (gk + 2 < K) t.z = ap[2];
                    if (gk + 3 < K) t.w = ap[3];
                }
                ra[r] = t;
            }
        }
    };
    // ---- issue global loads for the B tile at k0 into registers ----
    auto loadB = [&](int k0) {
        if (bT) {  // B stored N x K row-major
            const bool fast = nFull && (k0 + BK <= K);
            #pragma unroll
            for (int r = 0; r < 4; ++r) {
                const int gn = col0 + sm + (r << 5);
                const int gk = k0 + sk4;
                if (fast) {
                    rb[r] = *(const float4*)(B + (long long)gn * ldb + gk);
                } else {
                    float4 t = {0.f, 0.f, 0.f, 0.f};
                    if (gn < N) {
                        const float* bp = B + (long long)gn * ldb + gk;
                        if (gk     < K) t.x = bp[0];
                        if (gk + 1 < K) t.y = bp[1];
                        if (gk + 2 < K) t.z = bp[2];
                        if (gk + 3 < K) t.w = bp[3];
                    }
                    rb[r] = t;
                }
            }
        } else {   // B stored K x N row-major
            const bool fast = nFull && (k0 + BK <= K);
            #pragma unroll
            for (int r = 0; r < 4; ++r) {
                const int gk = k0 + skb + (r << 3);
                const int gn = col0 + snb;
                if (fast) {
                    rb[r] = *(const float4*)(B + (long long)gk * ldb + gn);
                } else {
                    float4 t = {0.f, 0.f, 0.f, 0.f};
                    if (gk < K) {
                        const float* bp = B + (long long)gk * ldb + gn;
                        if (gn     < N) t.x = bp[0];
                        if (gn + 1 < N) t.y = bp[1];
                        if (gn + 2 < N) t.z = bp[2];
                        if (gn + 3 < N) t.w = bp[3];
                    }
                    rb[r] = t;
                }
            }
        }
    };
    // ---- convert + commit register tiles to LDS (packed 64-bit stores) ----
    auto storeA = [&]() {
        #pragma unroll
        for (int r = 0; r < 4; ++r)
            *(v4bf*)&As[(sm + (r << 5)) * LDSS + sk4] = pack4_bf16(ra[r]);
    };
    auto storeB = [&]() {
        if (bT) {
            #pragma unroll
            for (int r = 0; r < 4; ++r)
                *(v4bf*)&Bs[(sm + (r << 5)) * LDSS + sk4] = pack4_bf16(rb[r]);
        } else {   // transpose into Bs[n][k]: scattered rows -> scalar stores
            #pragma unroll
            for (int r = 0; r < 4; ++r) {
                const int k = skb + (r << 3);
                Bs[(snb + 0) * LDSS + k] = (bf16_t)rb[r].x;
                Bs[(snb + 1) * LDSS + k] = (bf16_t)rb[r].y;
                Bs[(snb + 2) * LDSS + k] = (bf16_t)rb[r].z;
                Bs[(snb + 3) * LDSS + k] = (bf16_t)rb[r].w;
            }
        }
    };

    v8f acc[4][2];
    const v8f vzero = {0.f,0.f,0.f,0.f,0.f,0.f,0.f,0.f};
    #pragma unroll
    for (int i = 0; i < 4; ++i)
        #pragma unroll
        for (int j = 0; j < 2; ++j) acc[i][j] = vzero;

    // prologue: prefetch first tiles
    loadA(0);
    loadB(0);

    for (int k0 = 0; k0 < K; k0 += BK) {
        storeA();
        storeB();
        __syncthreads();

        // issue next tile's global loads; they complete while we WMMA
        const int kn = k0 + BK;
        if (kn < K) { loadA(kn); loadB(kn); }

        // ---- WMMA compute: fragments per CDNA5 wave32 layouts ----
        const int mlane = lane & 15;
        const int kg    = lane >> 4;   // K-group select
        v16bf afrag[4];
        #pragma unroll
        for (int i = 0; i < 4; ++i) {
            // A 16x32 bf16: lane -> row M; elems 0..7 = K(kg*8..kg*8+7),
            // elems 8..15 = K(16+kg*8..23+kg*8)
            const bf16_t* p = &As[(wm * 64 + i * 16 + mlane) * LDSS + kg * 8];
            v8bf lo = *(const v8bf*)p;
            v8bf hi = *(const v8bf*)(p + 16);
            afrag[i] = __builtin_shufflevector(lo, hi,
                0,1,2,3,4,5,6,7,8,9,10,11,12,13,14,15);
        }
        v16bf bfrag[2];
        #pragma unroll
        for (int j = 0; j < 2; ++j) {
            // B 32x16 bf16 (column-major staged): lane -> col N;
            // elems 0..7 = K(kg*8..kg*8+7), elems 8..15 = K(16+kg*8..23+kg*8)
            const bf16_t* p = &Bs[(wn * 32 + j * 16 + mlane) * LDSS + kg * 8];
            v8bf lo = *(const v8bf*)p;
            v8bf hi = *(const v8bf*)(p + 16);
            bfrag[j] = __builtin_shufflevector(lo, hi,
                0,1,2,3,4,5,6,7,8,9,10,11,12,13,14,15);
        }
        #pragma unroll
        for (int i = 0; i < 4; ++i)
            #pragma unroll
            for (int j = 0; j < 2; ++j)
                acc[i][j] = __builtin_amdgcn_wmma_f32_16x16x32_bf16(
                    false, afrag[i], false, bfrag[j],
                    (short)0, acc[i][j], false, false);
        __syncthreads();
    }

    // ---- epilogue: C layout -> VGPR r: M = r + 8*lane[4], N = lane[3:0] ----
    const int mlane = lane & 15;
    const int mhi   = (lane >> 4) * 8;
    if (mFull && nFull) {
        #pragma unroll
        for (int i = 0; i < 4; ++i) {
            #pragma unroll
            for (int j = 0; j < 2; ++j) {
                #pragma unroll
                for (int r = 0; r < 8; ++r) {
                    const int m = row0 + wm * 64 + i * 16 + mhi + r;
                    const int n = col0 + wn * 32 + j * 16 + mlane;
                    float vacc = acc[i][j][r];
                    if (bias) vacc += bias[n];
                    if (act)  vacc = gelu_erf(vacc);
                    if (Res)  vacc += Res[(long long)m * ldr + n];
                    C[(long long)m * ldc + n] = vacc;
                }
            }
        }
    } else {
        #pragma unroll
        for (int i = 0; i < 4; ++i) {
            #pragma unroll
            for (int j = 0; j < 2; ++j) {
                #pragma unroll
                for (int r = 0; r < 8; ++r) {
                    const int m = row0 + wm * 64 + i * 16 + mhi + r;
                    const int n = col0 + wn * 32 + j * 16 + mlane;
                    if (m < M && n < N) {
                        float vacc = acc[i][j][r];
                        if (bias) vacc += bias[n];
                        if (act)  vacc = gelu_erf(vacc);
                        if (Res)  vacc += Res[(long long)m * ldr + n];
                        C[(long long)m * ldc + n] = vacc;
                    }
                }
            }
        }
    }
}

// ---------------------------------------------------------------------------
// Embedding + sinusoidal positional encoding
// ---------------------------------------------------------------------------
__global__ void embed_kernel(const int* __restrict__ tok,
                             const float* __restrict__ emb,
                             float* __restrict__ out, int S, int d_model)
{
    const long long idx = (long long)blockIdx.x * blockDim.x + threadIdx.x;
    const long long total = (long long)Bb * S * d_model;
    if (idx >= total) return;
    const int d = (int)(idx % d_model);
    const int s = (int)((idx / d_model) % S);
    const int b = (int)(idx / ((long long)S * d_model));
    const int t = tok[b * S + s];
    const int i2 = d & ~1;
    const float freq = __expf(-(float)i2 * (9.210340371976184f / (float)d_model));
    const float ang  = (float)s * freq;
    const float pe   = (d & 1) ? __cosf(ang) : __sinf(ang);
    out[idx] = emb[(long long)t * d_model + d] * sqrtf((float)d_model) + pe;
}

// ---------------------------------------------------------------------------
// LayerNorm over last dim (one block per row)
// ---------------------------------------------------------------------------
__global__ void layernorm_rows(const float* __restrict__ x,
                               const float* __restrict__ g,
                               const float* __restrict__ b,
                               float* __restrict__ out, int d)
{
    const long long row = blockIdx.x;
    x   += row * d;
    out += row * d;
    __shared__ float red[256];
    float s = 0.f;
    for (int i = threadIdx.x; i < d; i += 256) s += x[i];
    red[threadIdx.x] = s; __syncthreads();
    for (int k = 128; k > 0; k >>= 1) {
        if (threadIdx.x < k) red[threadIdx.x] += red[threadIdx.x + k];
        __syncthreads();
    }
    const float mean = red[0] / (float)d;
    __syncthreads();
    float vs = 0.f;
    for (int i = threadIdx.x; i < d; i += 256) {
        const float dv = x[i] - mean;
        vs += dv * dv;
    }
    red[threadIdx.x] = vs; __syncthreads();
    for (int k = 128; k > 0; k >>= 1) {
        if (threadIdx.x < k) red[threadIdx.x] += red[threadIdx.x + k];
        __syncthreads();
    }
    const float rstd = rsqrtf(red[0] / (float)d + 1e-5f);
    for (int i = threadIdx.x; i < d; i += 256)
        out[i] = (x[i] - mean) * rstd * g[i] + b[i];
}

// ---------------------------------------------------------------------------
// In-place row softmax / log-softmax (one block per row)
// ---------------------------------------------------------------------------
__global__ void softmax_rows(float* __restrict__ x, int len)
{
    const long long row = blockIdx.x;
    x += row * len;
    __shared__ float red[256];
    float m = -3.4e38f;
    for (int i = threadIdx.x; i < len; i += 256) m = fmaxf(m, x[i]);
    red[threadIdx.x] = m; __syncthreads();
    for (int k = 128; k > 0; k >>= 1) {
        if (threadIdx.x < k) red[threadIdx.x] = fmaxf(red[threadIdx.x], red[threadIdx.x + k]);
        __syncthreads();
    }
    m = red[0]; __syncthreads();
    float s = 0.f;
    for (int i = threadIdx.x; i < len; i += 256) {
        const float e = __expf(x[i] - m);
        x[i] = e; s += e;
    }
    red[threadIdx.x] = s; __syncthreads();
    for (int k = 128; k > 0; k >>= 1) {
        if (threadIdx.x < k) red[threadIdx.x] += red[threadIdx.x + k];
        __syncthreads();
    }
    const float inv = 1.f / red[0];
    for (int i = threadIdx.x; i < len; i += 256) x[i] *= inv;
}

__global__ void logsoftmax_rows(float* __restrict__ x, int len)
{
    const long long row = blockIdx.x;
    x += row * len;
    __shared__ float red[256];
    float m = -3.4e38f;
    for (int i = threadIdx.x; i < len; i += 256) m = fmaxf(m, x[i]);
    red[threadIdx.x] = m; __syncthreads();
    for (int k = 128; k > 0; k >>= 1) {
        if (threadIdx.x < k) red[threadIdx.x] = fmaxf(red[threadIdx.x], red[threadIdx.x + k]);
        __syncthreads();
    }
    m = red[0]; __syncthreads();
    float s = 0.f;
    for (int i = threadIdx.x; i < len; i += 256) s += __expf(x[i] - m);
    red[threadIdx.x] = s; __syncthreads();
    for (int k = 128; k > 0; k >>= 1) {
        if (threadIdx.x < k) red[threadIdx.x] += red[threadIdx.x + k];
        __syncthreads();
    }
    const float lse = logf(red[0]) + m;
    for (int i = threadIdx.x; i < len; i += 256) x[i] -= lse;
}

// ---------------------------------------------------------------------------
// Host orchestration
// ---------------------------------------------------------------------------
extern "C" void kernel_launch(void* const* d_in, const int* in_sizes, int n_in,
                              void* d_out, int out_size, void* d_ws, size_t ws_size,
                              hipStream_t stream)
{
    (void)in_sizes; (void)n_in; (void)out_size; (void)ws_size;

    const int*   src        = (const int*)d_in[0];
    const int*   tgt        = (const int*)d_in[1];
    const float* emb        = (const float*)d_in[5];
    const float* enc_qkv_w  = (const float*)d_in[6];
    const float* enc_qkv_b  = (const float*)d_in[7];
    const float* enc_proj_w = (const float*)d_in[8];
    const float* enc_proj_b = (const float*)d_in[9];
    const float* enc_ff_w1  = (const float*)d_in[10];
    const float* enc_ff_b1  = (const float*)d_in[11];
    const float* enc_ff_w2  = (const float*)d_in[12];
    const float* enc_ff_b2  = (const float*)d_in[13];
    const float* enc_ln_g   = (const float*)d_in[14];
    const float* enc_ln_b   = (const float*)d_in[15];
    const float* enc_norm_g = (const float*)d_in[16];
    const float* enc_norm_b = (const float*)d_in[17];
    const float* dec_qkv_w  = (const float*)d_in[18];
    const float* dec_qkv_b  = (const float*)d_in[19];
    const float* dec_proj_w = (const float*)d_in[20];
    const float* dec_proj_b = (const float*)d_in[21];
    const float* dec_ff_w1  = (const float*)d_in[22];
    const float* dec_ff_b1  = (const float*)d_in[23];
    const float* dec_ff_w2  = (const float*)d_in[24];
    const float* dec_ff_b2  = (const float*)d_in[25];
    const float* dec_ln_g   = (const float*)d_in[26];
    const float* dec_ln_b   = (const float*)d_in[27];
    const float* dec_norm_g = (const float*)d_in[28];
    const float* dec_norm_b = (const float*)d_in[29];
    const float* gen_w      = (const float*)d_in[30];
    const float* gen_b      = (const float*)d_in[31];

    // ---- workspace carve-up (floats) ----
    float* ws = (float*)d_ws;
    size_t o = 0;
    float* x   = ws + o; o += (size_t)TOK * Dm;
    float* h   = ws + o; o += (size_t)TOK * Dm;
    float* q   = ws + o; o += (size_t)TOK * Dm;
    float* k   = ws + o; o += (size_t)TOK * Dm;
    float* v   = ws + o; o += (size_t)TOK * Dm;
    float* ao  = ws + o; o += (size_t)TOK * Dm;
    float* mem = ws + o; o += (size_t)TOK * Dm;
    float* y   = ws + o; o += (size_t)TOK * Dm;
    float* ffh = ws + o; o += (size_t)TOK * HFF;
    float* sc  = ws + o; o += (size_t)Bb * NHh * Ss * Ss;

    auto gemm = [&](const float* A, int lda, long long sA1, long long sA2,
                    const float* Bm, int ldb, long long sB1, long long sB2, int bT,
                    const float* bias, const float* Res, int ldr,
                    float* Cm, int ldc, long long sC1, long long sC2,
                    int M, int N, int K, int batches, int nh, int act) {
        dim3 grid((N + BN - 1) / BN, (M + BM - 1) / BM, batches);
        gemm_bf16_wmma<<<grid, 256, 0, stream>>>(
            A, lda, sA1, sA2, Bm, ldb, sB1, sB2, bT,
            bias, Res, ldr, Cm, ldc, sC1, sC2, M, N, K, nh, act);
    };
    // plain dense layer: C = act(A@W + bias) [+ Res]
    auto dense = [&](const float* A, const float* W, const float* bias,
                     const float* Res, float* Cm, int M, int N, int K, int act) {
        gemm(A, K, 0, 0, W, N, 0, 0, /*bT=*/0, bias, Res, N,
             Cm, N, 0, 0, M, N, K, 1, 1, act);
    };
    auto ln = [&](const float* xin, const float* g, const float* b, float* out) {
        layernorm_rows<<<TOK, 256, 0, stream>>>(xin, g, b, out, Dm);
    };
    // multi-head attention: io = io + proj(attn(qin, kvin))
    auto mha = [&](const float* qin, const float* kvin,
                   const float* wqkv, const float* bqkv,
                   const float* wo, const float* bo, float* io) {
        dense(qin,  wqkv + 0 * (size_t)Dm * Dm, bqkv + 0 * Dm, nullptr, q, TOK, Dm, Dm, 0);
        dense(kvin, wqkv + 1 * (size_t)Dm * Dm, bqkv + 1 * Dm, nullptr, k, TOK, Dm, Dm, 0);
        dense(kvin, wqkv + 2 * (size_t)Dm * Dm, bqkv + 2 * Dm, nullptr, v, TOK, Dm, Dm, 0);
        const long long sSD = (long long)Ss * Dm;
        const long long sSS = (long long)Ss * Ss;
        // scores[b,h] = Q[b,h] @ K[b,h]^T   (no scaling — faithful to source)
        gemm(q, Dm, DH, sSD,
             k, Dm, DH, sSD, /*bT=*/1, nullptr, nullptr, 0,
             sc, Ss, sSS, (long long)NHh * sSS,
             Ss, Ss, DH, Bb * NHh, NHh, 0);
        softmax_rows<<<Bb * NHh * Ss, 256, 0, stream>>>(sc, Ss);
        // out[b,h] = attn[b,h] @ V[b,h]
        gemm(sc, Ss, sSS, (long long)NHh * sSS,
             v, Dm, DH, sSD, /*bT=*/0, nullptr, nullptr, 0,
             ao, Dm, DH, sSD,
             Ss, DH, Ss, Bb * NHh, NHh, 0);
        // io = io + ao @ Wo + bo
        dense(ao, wo, bo, io, io, TOK, Dm, Dm, 0);
    };

    // ================= encoder =================
    {
        const long long n = (long long)TOK * Dm;
        embed_kernel<<<(unsigned)((n + 255) / 256), 256, 0, stream>>>(src, emb, x, Ss, Dm);
    }
    for (int i = 0; i < LEe; ++i) {
        ln(x, enc_ln_g + (size_t)(i * 2 + 0) * Dm, enc_ln_b + (size_t)(i * 2 + 0) * Dm, h);
        mha(h, h,
            enc_qkv_w + (size_t)i * 3 * Dm * Dm, enc_qkv_b + (size_t)i * 3 * Dm,
            enc_proj_w + (size_t)i * Dm * Dm,    enc_proj_b + (size_t)i * Dm, x);
        ln(x, enc_ln_g + (size_t)(i * 2 + 1) * Dm, enc_ln_b + (size_t)(i * 2 + 1) * Dm, h);
        dense(h,   enc_ff_w1 + (size_t)i * Dm * HFF, enc_ff_b1 + (size_t)i * HFF,
              nullptr, ffh, TOK, HFF, Dm, /*gelu*/1);
        dense(ffh, enc_ff_w2 + (size_t)i * HFF * Dm, enc_ff_b2 + (size_t)i * Dm,
              x, x, TOK, Dm, HFF, /*gelu*/1);
    }
    ln(x, enc_norm_g, enc_norm_b, mem);

    // ================= decoder =================
    {
        const long long n = (long long)TOK * Dm;
        embed_kernel<<<(unsigned)((n + 255) / 256), 256, 0, stream>>>(tgt, emb, y, Ss, Dm);
    }
    for (int i = 0; i < LDd; ++i) {
        // self-attention
        ln(y, dec_ln_g + (size_t)(i * 3 + 0) * Dm, dec_ln_b + (size_t)(i * 3 + 0) * Dm, h);
        mha(h, h,
            dec_qkv_w + (size_t)(i * 2 + 0) * 3 * Dm * Dm, dec_qkv_b + (size_t)(i * 2 + 0) * 3 * Dm,
            dec_proj_w + (size_t)(i * 2 + 0) * Dm * Dm,    dec_proj_b + (size_t)(i * 2 + 0) * Dm, y);
        // cross-attention
        ln(y, dec_ln_g + (size_t)(i * 3 + 1) * Dm, dec_ln_b + (size_t)(i * 3 + 1) * Dm, h);
        mha(h, mem,
            dec_qkv_w + (size_t)(i * 2 + 1) * 3 * Dm * Dm, dec_qkv_b + (size_t)(i * 2 + 1) * 3 * Dm,
            dec_proj_w + (size_t)(i * 2 + 1) * Dm * Dm,    dec_proj_b + (size_t)(i * 2 + 1) * Dm, y);
        // feed-forward
        ln(y, dec_ln_g + (size_t)(i * 3 + 2) * Dm, dec_ln_b + (size_t)(i * 3 + 2) * Dm, h);
        dense(h,   dec_ff_w1 + (size_t)i * Dm * HFF, dec_ff_b1 + (size_t)i * HFF,
              nullptr, ffh, TOK, HFF, Dm, 1);
        dense(ffh, dec_ff_w2 + (size_t)i * HFF * Dm, dec_ff_b2 + (size_t)i * Dm,
              y, y, TOK, Dm, HFF, 1);
    }
    ln(y, dec_norm_g, dec_norm_b, h);

    // ================= generator + log-softmax =================
    dense(h, gen_w, gen_b, nullptr, (float*)d_out, TOK, Vv, Dm, 0);
    logsoftmax_rows<<<TOK, 256, 0, stream>>>((float*)d_out, Vv);
}
